// Unified_Cross_Entropy_Loss_19344532701365
// MI455X (gfx1250) — matrix operand order repaired and checked by reading, not compiled, not run on hardware
//
#include <hip/hip_runtime.h>
#include <hip/hip_bf16.h>

typedef __attribute__((ext_vector_type(16))) __bf16 v16bf;
typedef __attribute__((ext_vector_type(8)))  float  v8f;
typedef __attribute__((ext_vector_type(4)))  unsigned int u32x4;
typedef __attribute__((ext_vector_type(8)))  int         i32x8;
typedef __attribute__((ext_vector_type(4)))  int         i32x4;

#define SCALE_S 64.0f
#define MARGIN_M 0.4f
#define EPS_N 1e-5f

#define BM 128
#define BN 128
#define KB 32
#define ASTR 40   // padded LDS row stride (shorts) -> 80B; matches TDM pad 16dw+4dw
#define BSTR 40

__device__ __forceinline__ unsigned short f32_to_bf16_rne(float f) {
  unsigned u = __float_as_uint(f);
  unsigned r = u + 0x7FFFu + ((u >> 16) & 1u);
  return (unsigned short)(r >> 16);
}

// ---------------------------------------------------------------------------
// Kernel 1: normalize each input row (f32) and store as bf16 into workspace.
// One 256-thread block per row (D == 512).
// ---------------------------------------------------------------------------
__global__ __launch_bounds__(256) void k_norm_input(const float* __restrict__ in,
                                                    unsigned short* __restrict__ abf,
                                                    int D) {
  const int row = blockIdx.x;
  const int t = threadIdx.x;
  const float* rp = in + (size_t)row * D;
  float x0 = rp[t];
  float x1 = rp[t + 256];
  float ss = x0 * x0 + x1 * x1;
  #pragma unroll
  for (int off = 16; off; off >>= 1) ss += __shfl_xor(ss, off, 32);
  __shared__ float red[8];
  const int wid = t >> 5, lane = t & 31;
  if (lane == 0) red[wid] = ss;
  __syncthreads();
  float tot = 0.f;
  #pragma unroll
  for (int i = 0; i < 8; ++i) tot += red[i];
  const float inv = 1.0f / fmaxf(sqrtf(tot), EPS_N);
  unsigned short* op = abf + (size_t)row * D;
  op[t]       = f32_to_bf16_rne(x0 * inv);
  op[t + 256] = f32_to_bf16_rne(x1 * inv);
}

// ---------------------------------------------------------------------------
// Kernel 2: per-weight-row inverse norm: invwn[c] = 1/max(||w_c||, eps).
// One wave per row; 8 rows per 256-thread block (D == 512).
// ---------------------------------------------------------------------------
__global__ __launch_bounds__(256) void k_wnorm(const float* __restrict__ w,
                                               float* __restrict__ invwn,
                                               int C, int D) {
  const int wid = threadIdx.x >> 5, lane = threadIdx.x & 31;
  const int c = blockIdx.x * 8 + wid;
  if (c >= C) return;  // uniform per wave
  const float4* rp = (const float4*)(w + (size_t)c * D) + lane * 4;
  float ss = 0.f;
  #pragma unroll
  for (int i = 0; i < 4; ++i) {
    float4 v = rp[i];
    ss += v.x * v.x + v.y * v.y + v.z * v.z + v.w * v.w;
  }
  #pragma unroll
  for (int off = 16; off; off >>= 1) ss += __shfl_xor(ss, off, 32);
  if (lane == 0) invwn[c] = 1.0f / fmaxf(sqrtf(ss), EPS_N);
}

// ---------------------------------------------------------------------------
// Kernel 3: tiled bf16 WMMA GEMM fused with the unified-cross-entropy loss
// epilogue and global reduction. Block tile 128x128, 8 waves, wave tile 64x32.
// A tile is DMA'd into LDS by the Tensor Data Mover (with HW padding matching
// ASTR); W tile is loaded f32 and converted to bf16 in-flight.
// ---------------------------------------------------------------------------
__global__ __launch_bounds__(256) void k_uce_main(
    const unsigned short* __restrict__ abf,   // [B,D] bf16 normalized input
    const float* __restrict__ w,              // [C,D] f32 raw weight
    const float* __restrict__ invwn,          // [C]
    const long long* __restrict__ label,      // [B]
    const float* __restrict__ bias,           // [1]
    float* __restrict__ out,                  // scalar
    int B, int C, int D) {
  __shared__ __align__(16) unsigned short At[BM * ASTR];
  __shared__ __align__(16) unsigned short Bt[BN * BSTR];
  __shared__ long long labs[BM];
  __shared__ float red[8];

  const int t = threadIdx.x;
  const int lane = t & 31;
  const int wid = t >> 5;
  const int waveM = wid >> 2;  // 0..1
  const int waveN = wid & 3;   // 0..3
  const int b0 = blockIdx.y * BM;
  const int c0 = blockIdx.x * BN;
  const int mrow = lane & 15;
  const int hi = lane >> 4;

  if (t < BM) labs[t] = label[b0 + t];
  const float bias_v = bias[0];

  const v8f zero = {};
  v8f acc[4][2];
  #pragma unroll
  for (int i = 0; i < 4; ++i)
    #pragma unroll
    for (int j = 0; j < 2; ++j) acc[i][j] = zero;

  union Frag { v16bf v; unsigned int u[8]; };

  // --- Tensor DMA descriptor pieces that are loop-invariant -----------------
  // group1: workgroup_mask=0; data_size=1 (2B); pad_enable; pad_interval=3
  // (16 DWORDs = 32 bf16 of payload); pad_amount=3 (4 DWORDs = 8 bf16) ->
  // effective LDS row stride 40 shorts == ASTR.
  i32x8 g1;
  g1[0] = (1 << 16) | (1 << 20) | (3 << 22) | (3 << 25);
  g1[1] = (int)((unsigned)(D & 0xFFFF) << 16);     // tensor_dim0[15:0]
  g1[2] = (int)((unsigned)(B & 0xFFFF) << 16);     // dim0 hi=0 | tensor_dim1[15:0]
  g1[3] = (int)(32u << 16);                        // dim1 hi=0 | tile_dim0=32
  g1[4] = BM;                                      // tile_dim1=128 | tile_dim2=0
  g1[5] = D;                                       // tensor_dim0_stride lo
  g1[6] = 0;                                       // stride hi | dim1_stride lo
  g1[7] = 0;
  const i32x4 gz  = {0, 0, 0, 0};
  const i32x8 gz8 = {0, 0, 0, 0, 0, 0, 0, 0};
  const unsigned lds_at = (unsigned)(unsigned long long)(uintptr_t)At;

  const int kIters = D / KB;  // 16
  for (int kt = 0; kt < kIters; ++kt) {
    const int k0 = kt * KB;
    __syncthreads();  // previous iteration's fragment reads done

    // --- A tile via Tensor Data Mover: 128x32 bf16 -> LDS (HW padded) ------
    if (wid == 0) {
      const unsigned long long ga =
          (unsigned long long)(uintptr_t)(abf + (size_t)b0 * D + k0);
      u32x4 g0;
      g0[0] = 1u;                                   // count=1, user descriptor
      g0[1] = lds_at;                               // lds_addr (bytes)
      g0[2] = (unsigned)(ga & 0xFFFFFFFFu);         // global_addr[31:0]
      g0[3] = (unsigned)((ga >> 32) & 0x01FFFFFFu) | (2u << 30);  // type=2
      __builtin_amdgcn_tensor_load_to_lds(g0, g1, gz, gz, gz8, 0);
    }

    // --- W tile: BN x KB f32 -> bf16 (8 KB in LDS), 4 chunks per thread ----
    #pragma unroll
    for (int i = 0; i < 4; ++i) {
      const int idx = t + i * 256;
      const int r = idx >> 3;
      const int cc = idx & 7;
      const int c = c0 + r;
      float4 v = make_float4(0.f, 0.f, 0.f, 0.f);
      if (c < C) v = *(const float4*)(w + (size_t)c * D + k0 + cc * 4);
      const unsigned long long pk =
          (unsigned long long)f32_to_bf16_rne(v.x) |
          ((unsigned long long)f32_to_bf16_rne(v.y) << 16) |
          ((unsigned long long)f32_to_bf16_rne(v.z) << 32) |
          ((unsigned long long)f32_to_bf16_rne(v.w) << 48);
      *(unsigned long long*)(Bt + r * BSTR + cc * 4) = pk;
    }
    // prefetch next K-slice of the weight tile into cache
    if (kt + 1 < kIters) {
      const int pr = t >> 1;
      if (c0 + pr < C)
        __builtin_prefetch(w + (size_t)(c0 + pr) * D + k0 + KB + (t & 1) * 16, 0, 1);
    }
    if (wid == 0) __builtin_amdgcn_s_wait_tensorcnt(0);  // A tile landed
    __syncthreads();

    // --- build fragments per ISA 7.12.2 layouts ----------------------------
    Frag fa[4], fb[2];
    #pragma unroll
    for (int ms = 0; ms < 4; ++ms) {
      const unsigned short* ap = At + (waveM * 64 + ms * 16 + mrow) * ASTR;
      #pragma unroll
      for (int v = 0; v < 8; ++v) {
        const int k = ((v & 4) << 2) + hi * 8 + ((v & 3) << 1);  // even
        fa[ms].u[v] = *(const unsigned int*)(ap + k);
      }
    }
    #pragma unroll
    for (int ns = 0; ns < 2; ++ns) {
      const unsigned short* bp = Bt + (waveN * 32 + ns * 16 + mrow) * BSTR;
      #pragma unroll
      for (int j = 0; j < 8; ++j) {
        const int k = ((j & 4) << 2) + hi * 8 + ((j & 3) << 1);  // even
        fb[ns].u[j] = *(const unsigned int*)(bp + k);
      }
    }
    // --- 8 WMMAs: D = A x B + C, f32 accumulate ---------------------------
    #pragma unroll
    for (int ms = 0; ms < 4; ++ms)
      #pragma unroll
      for (int ns = 0; ns < 2; ++ns)
        acc[ms][ns] = __builtin_amdgcn_wmma_f32_16x16x32_bf16(
            false, fa[ms].v, false, fb[ns].v, (short)0, acc[ms][ns],
            false, false);
  }

  // --- fused loss epilogue + reduction -------------------------------------
  float lsum = 0.f;
  #pragma unroll
  for (int ns = 0; ns < 2; ++ns) {
    const int c = c0 + waveN * 32 + ns * 16 + mrow;
    if (c >= C) continue;
    const float iwn = invwn[c];
    #pragma unroll
    for (int ms = 0; ms < 4; ++ms) {
      #pragma unroll
      for (int r = 0; r < 8; ++r) {
        const int m = waveM * 64 + ms * 16 + hi * 8 + r;  // C/D layout: M = r + 8*hi
        const float cosv = acc[ms][ns][r] * iwn;
        const long long lab = labs[m];
        const float tn = fminf(fmaxf(SCALE_S * cosv - bias_v, -SCALE_S), SCALE_S);
        const float tp = fminf(fmaxf(SCALE_S * (cosv - MARGIN_M) - bias_v, -SCALE_S), SCALE_S);
        const float x = ((long long)c == lab) ? -tp : tn;
        lsum += log1pf(__expf(x));
      }
    }
  }
  #pragma unroll
  for (int off = 16; off; off >>= 1) lsum += __shfl_xor(lsum, off, 32);
  if (lane == 0) red[wid] = lsum;
  __syncthreads();
  if (t == 0) {
    float tot = 0.f;
    #pragma unroll
    for (int i = 0; i < 8; ++i) tot += red[i];
    atomicAdd(out, tot * (1.0f / (float)B));
  }
}

// ---------------------------------------------------------------------------
extern "C" void kernel_launch(void* const* d_in, const int* in_sizes, int n_in,
                              void* d_out, int out_size, void* d_ws, size_t ws_size,
                              hipStream_t stream) {
  const float*     inp    = (const float*)d_in[0];
  const long long* label  = (const long long*)d_in[1];
  const float*     weight = (const float*)d_in[2];
  const float*     bias   = (const float*)d_in[3];
  const int B = in_sizes[1];           // 512 labels
  const int D = in_sizes[0] / B;       // 512
  const int C = in_sizes[2] / D;       // 100000

  unsigned short* abf   = (unsigned short*)d_ws;                         // B*D bf16
  float*          invwn = (float*)((char*)d_ws +
                                   (size_t)B * D * sizeof(unsigned short));
  float* out = (float*)d_out;

  (void)hipMemsetAsync(d_out, 0, sizeof(float), stream);
  k_norm_input<<<B, 256, 0, stream>>>(inp, abf, D);
  k_wnorm<<<(C + 7) / 8, 256, 0, stream>>>(weight, invwn, C, D);
  dim3 grid((C + BN - 1) / BN, (B + BM - 1) / BM);
  k_uce_main<<<grid, 256, 0, stream>>>(abf, weight, invwn, label, bias, out,
                                       B, C, D);
}
